// ContextRGAT_90546500534351
// MI455X (gfx1250) — matrix operand order repaired and checked by reading, not compiled, not run on hardware
//
#include <hip/hip_runtime.h>
#include <hip/hip_bf16.h>
#include <math.h>
#include <stdint.h>

// ---------------------------------------------------------------------------
// RGAT (2 layers, heads=1, dim=1) for MI455X / gfx1250 (wave32, WMMA).
// N=50000 (=3125*16), E=800000, R=3, all feature dims = 128.
// Dense transform: bf16 WMMA with pre-packed operands + async-LDS-staged A.
// Edge phase: f32, memory-bound, atomic segment softmax + scatter.
// ---------------------------------------------------------------------------

#define NNODES   50000
#define NEDGES   800000
#define NREL     3
#define FDIM     128

typedef __attribute__((ext_vector_type(16))) __bf16 v16bf;
typedef __attribute__((ext_vector_type(8)))  float  v8f;

// Async global->LDS path (CDNA5, ASYNCcnt-tracked). Signature learned from
// the round-2 diagnostic: arg0 is AS(1) int4*, 4 args total.
#if defined(__AMDGCN__) && \
    __has_builtin(__builtin_amdgcn_global_load_async_to_lds_b128) && \
    __has_builtin(__builtin_amdgcn_s_wait_asynccnt)
#define USE_ASYNC_LDS 1
typedef int v4i_t __attribute__((__vector_size__(4 * sizeof(int))));
typedef __attribute__((address_space(1))) v4i_t* gas_ptr;
typedef __attribute__((address_space(3))) v4i_t* las_ptr;
#else
#define USE_ASYNC_LDS 0
#endif

// ---------------------------------------------------------------------------
// Pre-pass A: elementwise f32 -> bf16 (row-major X or H)
// ---------------------------------------------------------------------------
__global__ void rgat_cvt_x(const float* __restrict__ X,
                           __bf16* __restrict__ XA, int total)
{
    const int i = blockIdx.x * blockDim.x + threadIdx.x;
    if (i < total) XA[i] = (__bf16)X[i];
}

// ---------------------------------------------------------------------------
// Pre-pass B: pack W[r][k][o] (f32) into bf16 WMMA-B fragment order:
//   WB[(((r*8 + ct)*4 + ks)*32 + lane)*16 + e] = W[r][ks*32 + half*16 + e][ct*16 + n]
// with half = lane>>4, n = lane&15.  One thread per (r,ct,ks,lane) = 3072.
// ---------------------------------------------------------------------------
__global__ void rgat_cvt_w(const float* __restrict__ W, __bf16* __restrict__ WB)
{
    const int t = blockIdx.x * blockDim.x + threadIdx.x;
    if (t >= NREL * 8 * 4 * 32) return;
    const int lane = t & 31;
    const int ks   = (t >> 5) & 3;
    const int ct   = (t >> 7) & 7;
    const int r    = t >> 10;
    const int half = lane >> 4, n = lane & 15;

    const float* wsrc = W + ((size_t)r * FDIM + ks * 32 + half * 16) * FDIM
                          + ct * 16 + n;
    __bf16* dstp = WB + (size_t)t * 16;
    #pragma unroll
    for (int e = 0; e < 16; ++e)
        dstp[e] = (__bf16)wsrc[(size_t)e * FDIM];
}

// ---------------------------------------------------------------------------
// Kernel 1: per-relation transform  XW[n][r][o] = sum_k X[n][k]*W[r][k][o]
// Block = 8 waves = one 16-row slab, grid.y = relation.
// A tile (16x128 bf16, 4 KB) staged to LDS once (async-to-LDS if available),
// then per wave: 4x v_wmma_f32_16x16x32_bf16 over K=128.
// ---------------------------------------------------------------------------
__global__ __launch_bounds__(256)
void rgat_xw_wmma(const __bf16* __restrict__ XA,   // [N,128] bf16 row-major
                  const __bf16* __restrict__ WB,   // packed B fragments
                  float* __restrict__ XW)          // [N,R,128] f32
{
    __shared__ __align__(16) __bf16 lds_a[16 * FDIM];   // 4 KB

    const int tid  = threadIdx.x;
    const int wave = tid >> 5;
    const int lane = tid & 31;
    const int half = lane >> 4;
    const int m    = lane & 15;
    const int row0 = blockIdx.x * 16;
    const int r    = blockIdx.y;

    // ---- stage A tile: 256 threads x 16 bytes --------------------------
    {
        const int arow = tid >> 4;          // 0..15
        const int achk = tid & 15;          // 0..15 (16B chunks per row)
        const __bf16* gsrc = XA + (size_t)(row0 + arow) * FDIM + achk * 8;
        __bf16*       ldst = lds_a + arow * FDIM + achk * 8;
#if USE_ASYNC_LDS
        __builtin_amdgcn_global_load_async_to_lds_b128(
            (gas_ptr)(uintptr_t)gsrc, (las_ptr)(uintptr_t)ldst, 0, 0);
        __builtin_amdgcn_s_wait_asynccnt(0);
#else
        *(uint4*)ldst = *(const uint4*)gsrc;
#endif
        __syncthreads();
    }

    v8f acc = {};
    #pragma unroll
    for (int ks = 0; ks < 4; ++ks) {
        const int kk = ks * 32;
        // A fragment from LDS: two ds_load_b128 per lane
        v16bf a;
        *(uint4*)&a =
            *(const uint4*)(lds_a + (size_t)m * FDIM + kk + half * 16);
        *((uint4*)&a + 1) =
            *(const uint4*)(lds_a + (size_t)m * FDIM + kk + half * 16 + 8);
        // B fragment: contiguous 32B per lane, lane-major (fully coalesced)
        const __bf16* bp = WB + ((((size_t)r * 8 + wave) * 4 + ks) * 32 + lane) * 16;
        v16bf b;
        *(uint4*)&b       = *(const uint4*)bp;
        *((uint4*)&b + 1) = *(const uint4*)(bp + 8);

        acc = __builtin_amdgcn_wmma_f32_16x16x32_bf16(
                  false, a, false, b, (short)0, acc, false, false);
    }

    // D store, C/D layout: VGPR v -> row = row0 + v + 8*half, col = col0 + m
    const int col0 = wave * 16;
    #pragma unroll
    for (int v = 0; v < 8; ++v) {
        const int row = row0 + v + 8 * half;
        XW[((size_t)row * NREL + r) * FDIM + (col0 + m)] = acc[v];
    }
}

// ---------------------------------------------------------------------------
// Kernel 2: node attention logits  NQ[n*3+r] = xw[n][r]·q,  NK = xw[n][r]·k
// ---------------------------------------------------------------------------
__global__ __launch_bounds__(256)
void rgat_logits(const float* __restrict__ XW,
                 const float* __restrict__ q, const float* __restrict__ k,
                 float* __restrict__ NQ, float* __restrict__ NK, int total)
{
    const int gwave = (int)((blockIdx.x * blockDim.x + threadIdx.x) >> 5);
    const int lane  = threadIdx.x & 31;
    if (gwave >= total) return;

    const float* xv = XW + (size_t)gwave * FDIM;
    float sq = 0.f, sk = 0.f;
    #pragma unroll
    for (int i = 0; i < 4; ++i) {
        const int c = lane * 4 + i;
        const float x = xv[c];
        sq += x * q[c];
        sk += x * k[c];
    }
    #pragma unroll
    for (int off = 16; off > 0; off >>= 1) {
        sq += __shfl_xor(sq, off, 32);
        sk += __shfl_xor(sk, off, 32);
    }
    if (lane == 0) { NQ[gwave] = sq; NK[gwave] = sk; }
}

// ---------------------------------------------------------------------------
// Kernel 3: edge-attr scalar  cE = sum_o we[0][o] * e[o][0]
// ---------------------------------------------------------------------------
__global__ void rgat_ce(const float* __restrict__ we,
                        const float* __restrict__ ev, float* __restrict__ cE)
{
    __shared__ float s[FDIM];
    const int t = threadIdx.x;
    s[t] = we[t] * ev[t];
    __syncthreads();
    for (int off = FDIM / 2; off > 0; off >>= 1) {
        if (t < off) s[t] += s[t + off];
        __syncthreads();
    }
    if (t == 0) *cE = s[0];
}

// ---------------------------------------------------------------------------
// Kernel 4: init accumulator / softmax stats
// ---------------------------------------------------------------------------
__global__ void rgat_init(float* __restrict__ accum,
                          float* __restrict__ amax, float* __restrict__ denom)
{
    const int i      = blockIdx.x * blockDim.x + threadIdx.x;
    const int stride = gridDim.x * blockDim.x;
    for (int idx = i; idx < NNODES * FDIM; idx += stride) accum[idx] = 0.f;
    for (int idx = i; idx < NNODES; idx += stride) {
        amax[idx]  = -INFINITY;
        denom[idx] = 0.f;
    }
}

__device__ __forceinline__ void atomicMaxFloat(float* addr, float v)
{
    if (v >= 0.f) atomicMax((int*)addr, __float_as_int(v));
    else          atomicMin((unsigned int*)addr, __float_as_uint(v));
}

// ---------------------------------------------------------------------------
// Kernel 5: alpha = leaky_relu(NQ[dst,et] + NK[src,et] + edge_attr*cE), seg max
// ---------------------------------------------------------------------------
__global__ void rgat_alpha(const int* __restrict__ src, const int* __restrict__ dst,
                           const int* __restrict__ et,  const float* __restrict__ eattr,
                           const float* __restrict__ NQ, const float* __restrict__ NK,
                           const float* __restrict__ cE,
                           float* __restrict__ alpha, float* __restrict__ amax)
{
    const int e = blockIdx.x * blockDim.x + threadIdx.x;
    if (e >= NEDGES) return;
    const int s = src[e], d = dst[e], t = et[e];
    float a = NQ[d * NREL + t] + NK[s * NREL + t] + eattr[e] * cE[0];
    a = (a > 0.f) ? a : 0.2f * a;
    alpha[e] = a;
    atomicMaxFloat(&amax[d], a);
}

// ---------------------------------------------------------------------------
// Kernel 6: ex = exp(alpha - amax[dst]) (in place), segment-sum denom
// ---------------------------------------------------------------------------
__global__ void rgat_exp(const int* __restrict__ dst,
                         float* __restrict__ alpha,
                         const float* __restrict__ amax,
                         float* __restrict__ denom)
{
    const int e = blockIdx.x * blockDim.x + threadIdx.x;
    if (e >= NEDGES) return;
    const int d = dst[e];
    const float ex = __expf(alpha[e] - amax[d]);
    alpha[e] = ex;
    atomicAdd(&denom[d], ex);
}

// ---------------------------------------------------------------------------
// Kernel 7: message scatter  accum[dst] += (ex/denom[dst]) * xw[src,et,:]
// One wave per edge; lane handles a float4 slice of the 128-wide message.
// ---------------------------------------------------------------------------
__global__ __launch_bounds__(256)
void rgat_msg(const int* __restrict__ src, const int* __restrict__ dst,
              const int* __restrict__ et,  const float* __restrict__ ex,
              const float* __restrict__ denom,
              const float* __restrict__ XW, float* __restrict__ accum)
{
    const int gwave = (int)((blockIdx.x * blockDim.x + threadIdx.x) >> 5);
    const int lane  = threadIdx.x & 31;
    if (gwave >= NEDGES) return;

    const int s = src[gwave], d = dst[gwave], t = et[gwave];
    const float attn = ex[gwave] / (denom[d] + 1e-16f);

    const float4 v = ((const float4*)(XW + ((size_t)s * NREL + t) * FDIM))[lane];
    float* outp = accum + (size_t)d * FDIM + lane * 4;
    atomicAdd(outp + 0, attn * v.x);
    atomicAdd(outp + 1, attn * v.y);
    atomicAdd(outp + 2, attn * v.z);
    atomicAdd(outp + 3, attn * v.w);
}

// ---------------------------------------------------------------------------
// Kernel 8: out = (accum + bias) [optionally relu], in place allowed
// ---------------------------------------------------------------------------
__global__ void rgat_bias_act(const float* __restrict__ accum,
                              const float* __restrict__ bias,
                              float* __restrict__ out, int do_relu)
{
    const int i = blockIdx.x * blockDim.x + threadIdx.x;
    if (i >= NNODES * FDIM) return;
    float v = accum[i] + bias[i & (FDIM - 1)];
    if (do_relu) v = v > 0.f ? v : 0.f;
    out[i] = v;
}

// ---------------------------------------------------------------------------
// Host launcher
// ---------------------------------------------------------------------------
extern "C" void kernel_launch(void* const* d_in, const int* in_sizes, int n_in,
                              void* d_out, int out_size, void* d_ws, size_t ws_size,
                              hipStream_t stream)
{
    (void)in_sizes; (void)n_in; (void)out_size; (void)ws_size;

    const float* x     = (const float*)d_in[0];
    const int*   eidx  = (const int*)  d_in[1];
    const int*   etype = (const int*)  d_in[2];
    const float* eattr = (const float*)d_in[3];
    const float* w1 = (const float*)d_in[4];
    const float* q1 = (const float*)d_in[5];
    const float* k1 = (const float*)d_in[6];
    const float* e1 = (const float*)d_in[7];
    const float* we1= (const float*)d_in[8];
    const float* b1 = (const float*)d_in[9];
    const float* w2 = (const float*)d_in[10];
    const float* q2 = (const float*)d_in[11];
    const float* k2 = (const float*)d_in[12];
    const float* e2 = (const float*)d_in[13];
    const float* we2= (const float*)d_in[14];
    const float* b2 = (const float*)d_in[15];

    const int* src = eidx;
    const int* dst = eidx + NEDGES;

    // ---- workspace layout ----------------------------------------------
    float* ws = (float*)d_ws;
    float* XW    = ws;                                   // N*R*128 f32
    float* H     = XW    + (size_t)NNODES * NREL * FDIM; // N*128 f32
    float* NQ    = H     + (size_t)NNODES * FDIM;        // N*R
    float* NK    = NQ    + (size_t)NNODES * NREL;        // N*R
    float* AMAX  = NK    + (size_t)NNODES * NREL;        // N
    float* DEN   = AMAX  + (size_t)NNODES;               // N
    float* ALPHA = DEN   + (size_t)NNODES;               // E
    float* CE    = ALPHA + (size_t)NEDGES;               // 1 (+3 pad)
    __bf16* XA   = (__bf16*)(CE + 4);                    // N*128 bf16 (16B aligned)
    __bf16* WB   = XA + (size_t)NNODES * FDIM;           // R*128*128 bf16

    // ---- launch geometry ------------------------------------------------
    const dim3 gemm_grid(NNODES / 16, NREL);                // 3125 x 3, exact
    const int  cvtx_blocks  = (NNODES * FDIM + 255) / 256;  // 25000
    const int  cvtw_blocks  = (NREL * 8 * 4 * 32 + 255) / 256; // 12
    const int  logit_blocks = (NNODES * NREL * 32) / 256;   // 18750, exact
    const int  node_blocks  = (NNODES * FDIM + 255) / 256;  // 25000
    const int  edge_blocks  = (NEDGES + 255) / 256;         // 3125
    const int  msg_blocks   = (NEDGES * 32) / 256;          // 100000, exact

    // =================== Layer 1 (x -> H, relu) =========================
    rgat_cvt_x  <<<cvtx_blocks, 256, 0, stream>>>(x, XA, NNODES * FDIM);
    rgat_cvt_w  <<<cvtw_blocks, 256, 0, stream>>>(w1, WB);
    rgat_xw_wmma<<<gemm_grid, 256, 0, stream>>>(XA, WB, XW);
    rgat_logits <<<logit_blocks, 256, 0, stream>>>(XW, q1, k1, NQ, NK, NNODES * NREL);
    rgat_ce     <<<1, FDIM, 0, stream>>>(we1, e1, CE);
    rgat_init   <<<node_blocks, 256, 0, stream>>>(H, AMAX, DEN);
    rgat_alpha  <<<edge_blocks, 256, 0, stream>>>(src, dst, etype, eattr,
                                                  NQ, NK, CE, ALPHA, AMAX);
    rgat_exp    <<<edge_blocks, 256, 0, stream>>>(dst, ALPHA, AMAX, DEN);
    rgat_msg    <<<msg_blocks, 256, 0, stream>>>(src, dst, etype, ALPHA, DEN, XW, H);
    rgat_bias_act<<<node_blocks, 256, 0, stream>>>(H, b1, H, /*relu=*/1);

    // =================== Layer 2 (H -> d_out) ===========================
    float* OUT = (float*)d_out;
    rgat_cvt_x  <<<cvtx_blocks, 256, 0, stream>>>(H, XA, NNODES * FDIM);
    rgat_cvt_w  <<<cvtw_blocks, 256, 0, stream>>>(w2, WB);
    rgat_xw_wmma<<<gemm_grid, 256, 0, stream>>>(XA, WB, XW);
    rgat_logits <<<logit_blocks, 256, 0, stream>>>(XW, q2, k2, NQ, NK, NNODES * NREL);
    rgat_ce     <<<1, FDIM, 0, stream>>>(we2, e2, CE);
    rgat_init   <<<node_blocks, 256, 0, stream>>>(OUT, AMAX, DEN);
    rgat_alpha  <<<edge_blocks, 256, 0, stream>>>(src, dst, etype, eattr,
                                                  NQ, NK, CE, ALPHA, AMAX);
    rgat_exp    <<<edge_blocks, 256, 0, stream>>>(dst, ALPHA, AMAX, DEN);
    rgat_msg    <<<msg_blocks, 256, 0, stream>>>(src, dst, etype, ALPHA, DEN, XW, OUT);
    rgat_bias_act<<<node_blocks, 256, 0, stream>>>(OUT, b2, OUT, /*relu=*/0);
}
